// Mamba2Block_87256555586261
// MI455X (gfx1250) — matrix operand order repaired and checked by reading, not compiled, not run on hardware
//
#include <hip/hip_runtime.h>
#include <hip/hip_bf16.h>

typedef __attribute__((ext_vector_type(16))) _Float16 v16h;
typedef __attribute__((ext_vector_type(8)))  float    v8f;

#define TOKENS   8192     // BATCH * L
#define LSEQ     4096
#define DMODEL   1024
#define DINNER   2048
#define DSTATE   16
#define DTRANK   64
#define NPROJ    96       // DT_RANK + 2*D_STATE

// chunked scan geometry
#define NCHUNK   16
#define CLEN     256      // LSEQ / NCHUNK

// ---------------- LDS tile geometry for WMMA GEMM ----------------
#define TILE_M   128
#define TILE_N   128
#define TILE_K   32
#define LDS_PAD  40       // halfs per row: 80B -> 16B-aligned rows, bank stride 20

// ---------------------------------------------------------------
// CDNA5 async tensor path: DMA global -> LDS with no data VGPRs (ASYNCcnt).
// LDS operand = low 32 bits of the flat shared pointer (ISA 10.2: for the LDS
// aperture, addr[31:0] is the workgroup-relative LDS address).
__device__ __forceinline__ void async_ld_b128(void* lds, const void* g) {
    unsigned off32 = (unsigned)(unsigned long long)lds;
    asm volatile("global_load_async_to_lds_b128 %0, %1, off"
                 :: "v"(off32), "v"(g) : "memory");
}
__device__ __forceinline__ void wait_async0() {
    asm volatile("s_wait_asynccnt 0x0" ::: "memory");
}

// ---------------------------------------------------------------
// Fragment loaders (per ISA 7.12.2 layouts)
// A 16x32 f16: lanes 0-15 -> M=lane, dwords {K0..7, K16..23}; lanes 16-31 -> {K8..15, K24..31}
__device__ __forceinline__ v16h load_a_frag(const _Float16* s, int lane) {
    int m  = lane & 15;
    int kd = (lane >> 4) * 4;                 // dword offset inside a row
    const unsigned int* p = (const unsigned int*)(s + m * LDS_PAD);
    union { v16h v; unsigned int u[8]; } r;
#pragma unroll
    for (int i = 0; i < 4; ++i) { r.u[i] = p[kd + i]; r.u[4 + i] = p[8 + kd + i]; }
    return r.v;
}
// B 32x16 f16 from [N,K] row-major: lanes 0-15 -> N=lane, {K0..7, K16..23}; lanes 16-31 -> {K8..15, K24..31}
__device__ __forceinline__ v16h load_b_frag(const _Float16* s, int lane) {
    int n  = lane & 15;
    int kd = (lane >> 4) * 4;
    const unsigned int* p = (const unsigned int*)(s + n * LDS_PAD);
    union { v16h v; unsigned int u[8]; } r;
#pragma unroll
    for (int i = 0; i < 4; ++i) { r.u[i] = p[kd + i]; r.u[4 + i] = p[8 + kd + i]; }
    return r.v;
}

__device__ __forceinline__ float softplus_f(float v) {
    return (v > 20.f) ? v : log1pf(__expf(v));
}

// ---------------------------------------------------------------
// Generic WMMA GEMM:  C[M,N](f32) = A[M,K](f16) @ B[N,K](f16)^T + bias (+ epilogue)
// Block tile 128x128, 8 waves as 2(M)x4(N); each wave: 4x2 WMMA tiles (64x32 output).
// LDS double-buffering via ASYNC loads: tile k+1 DMAs straight into buf^1 with
// zero staging VGPRs while the WMMA burst consumes buf; s_wait_asynccnt + one
// barrier per k-step.
// EPI 0: +bias (bias may be null)   EPI 1: softplus(acc+bias)   EPI 2: acc+bias+res
// GN: guard B rows / C cols against N (only needed when N % TILE_N != 0).
// M multiple of 128, K multiple of 32.
template <int EPI, bool GN>
__global__ __launch_bounds__(256, 1) __attribute__((amdgpu_waves_per_eu(1)))
void gemm_wmma_f16(const _Float16* __restrict__ A, const _Float16* __restrict__ Bw,
                   const float* __restrict__ bias, const float* __restrict__ res,
                   float* __restrict__ C, int M, int N, int K)
{
    __shared__ __align__(16) _Float16 sA[2][TILE_M * LDS_PAD];   // 2 x 10240 B
    __shared__ __align__(16) _Float16 sB[2][TILE_N * LDS_PAD];

    const int tid  = threadIdx.x;
    const int lane = tid & 31;
    const int wave = tid >> 5;
    const int m0 = blockIdx.y * TILE_M;
    const int n0 = blockIdx.x * TILE_N;
    const int wm = (wave >> 2) * 64;          // 2 waves along M, 64 rows each
    const int wn = (wave & 3) * 32;           // 4 waves along N, 32 cols each

    // staging chunk mapping: 128 rows x 32 halfs = 512 uint4 chunks, 2 per thread
    int srow[2], scol[2];
#pragma unroll
    for (int c = 0; c < 2; ++c) {
        int ch = tid + c * 256;
        srow[c] = ch >> 2;
        scol[c] = (ch & 3) * 8;
    }

    v8f acc[4][2] = {};

    // GN case: B rows >= N are never DMA-written; zero them once in both buffers
    if (GN) {
#pragma unroll
        for (int c = 0; c < 2; ++c) {
            if (n0 + srow[c] >= N) {
                uint4 z = {0u, 0u, 0u, 0u};
                *(uint4*)(sB[0] + srow[c] * LDS_PAD + scol[c]) = z;
                *(uint4*)(sB[1] + srow[c] * LDS_PAD + scol[c]) = z;
            }
        }
    }

    auto async_tile = [&](int buf, int kk) {
#pragma unroll
        for (int c = 0; c < 2; ++c) {
            async_ld_b128(sA[buf] + srow[c] * LDS_PAD + scol[c],
                          A + (size_t)(m0 + srow[c]) * K + kk + scol[c]);
            if (!GN || (n0 + srow[c] < N))
                async_ld_b128(sB[buf] + srow[c] * LDS_PAD + scol[c],
                              Bw + (size_t)(n0 + srow[c]) * K + kk + scol[c]);
        }
    };

    // prologue: DMA tile 0 into buf 0
    async_tile(0, 0);
    wait_async0();
    __syncthreads();

    int buf = 0;
    for (int k0 = 0; k0 < K; k0 += TILE_K) {
        const int k1 = k0 + TILE_K;
        if (k1 < K) {
            async_tile(buf ^ 1, k1);          // DMA next tile while computing
            if (k1 + TILE_K < K) {            // warm L2/WGP$ two tiles ahead
                int prow = tid >> 1;
                int pcol = (tid & 1) * 16;
                __builtin_prefetch(A + (size_t)(m0 + prow) * K + k1 + TILE_K + pcol, 0, 1);
                if (!GN || (n0 + prow < N))
                    __builtin_prefetch(Bw + (size_t)(n0 + prow) * K + k1 + TILE_K + pcol, 0, 1);
            }
        }

        // WMMA burst: B-fragments reused across 4 M-tiles, A-fragments across 2 N-tiles
        v16h fb0 = load_b_frag(sB[buf] + wn * LDS_PAD, lane);
        v16h fb1 = load_b_frag(sB[buf] + (wn + 16) * LDS_PAD, lane);
#pragma unroll
        for (int mi = 0; mi < 4; ++mi) {
            v16h fa = load_a_frag(sA[buf] + (wm + mi * 16) * LDS_PAD, lane);
            acc[mi][0] = __builtin_amdgcn_wmma_f32_16x16x32_f16(false, fa, false, fb0,
                                                               (short)0, acc[mi][0], false, false);
            acc[mi][1] = __builtin_amdgcn_wmma_f32_16x16x32_f16(false, fa, false, fb1,
                                                               (short)0, acc[mi][1], false, false);
        }

        wait_async0();                        // next tile has landed in buf^1
        __syncthreads();                      // one barrier per k-step
        buf ^= 1;
    }

    // epilogue: C/D layout -> lane 0-15: N=lane, M=r ; lane 16-31: N=lane-16, M=8+r
    const int mloc = (lane >> 4) * 8;
    const int nl   = lane & 15;
#pragma unroll
    for (int ni = 0; ni < 2; ++ni) {
        int gn = n0 + wn + ni * 16 + nl;
        if (!GN || gn < N) {
            float bvv = bias ? bias[gn] : 0.f;
#pragma unroll
            for (int mi = 0; mi < 4; ++mi) {
#pragma unroll
                for (int r = 0; r < 8; ++r) {
                    int gm = m0 + wm + mi * 16 + mloc + r;
                    float v = acc[mi][ni][r] + bvv;
                    if (EPI == 1) v = softplus_f(v);
                    if (EPI == 2) v += res[(size_t)gm * N + gn];
                    C[(size_t)gm * N + gn] = v;
                }
            }
        }
    }
}

// ---------------------------------------------------------------
// LayerNorm over D_MODEL=1024 + cast to f16. One block per token.
__global__ __launch_bounds__(256)
void ln_cast_kernel(const float* __restrict__ x, const float* __restrict__ w,
                    const float* __restrict__ b, _Float16* __restrict__ xn)
{
    __shared__ float red[256];
    const int tok = blockIdx.x, tid = threadIdx.x;
    const float* row = x + (size_t)tok * DMODEL;
    float v[4], s = 0.f;
#pragma unroll
    for (int i = 0; i < 4; ++i) { v[i] = row[tid + i * 256]; s += v[i]; }
    red[tid] = s; __syncthreads();
    for (int st = 128; st > 0; st >>= 1) { if (tid < st) red[tid] += red[tid + st]; __syncthreads(); }
    float mu = red[0] * (1.f / DMODEL);
    __syncthreads();
    float s2 = 0.f;
#pragma unroll
    for (int i = 0; i < 4; ++i) { float dv = v[i] - mu; s2 += dv * dv; }
    red[tid] = s2; __syncthreads();
    for (int st = 128; st > 0; st >>= 1) { if (tid < st) red[tid] += red[tid + st]; __syncthreads(); }
    float rstd = rsqrtf(red[0] * (1.f / DMODEL) + 1e-5f);
#pragma unroll
    for (int i = 0; i < 4; ++i) {
        int c = tid + i * 256;
        xn[(size_t)tok * DMODEL + c] = (_Float16)((v[i] - mu) * rstd * w[c] + b[c]);
    }
}

// f32 -> f16 cast
__global__ __launch_bounds__(256)
void cast_h_kernel(const float* __restrict__ src, _Float16* __restrict__ dst, int n)
{
    int i = blockIdx.x * 256 + threadIdx.x;
    if (i < n) dst[i] = (_Float16)src[i];
}

// dt columns (first 64 of proj[.,96]) -> f16
__global__ __launch_bounds__(256)
void extract_dt_kernel(const float* __restrict__ proj, _Float16* __restrict__ dt_h)
{
    int i = blockIdx.x * 256 + threadIdx.x;      // TOKENS*DTRANK
    int tok = i >> 6, r = i & 63;
    dt_h[i] = (_Float16)proj[(size_t)tok * NPROJ + r];
}

// causal depthwise conv (k=4) over time + SiLU; reads x_in half of xz
__global__ __launch_bounds__(256)
void conv_silu_kernel(const float* __restrict__ xz, const float* __restrict__ cw,
                      const float* __restrict__ cb, float* __restrict__ xc,
                      _Float16* __restrict__ xc_h)
{
    int idx = blockIdx.x * 256 + threadIdx.x;    // TOKENS * DINNER
    int d = idx & (DINNER - 1);
    int token = idx >> 11;
    int t = token & (LSEQ - 1);
    int bb = token >> 12;
    float acc = cb[d];
#pragma unroll
    for (int j = 0; j < 4; ++j) {
        int ts = t - 3 + j;
        if (ts >= 0)
            acc += cw[d * 4 + j] * xz[(size_t)(bb * LSEQ + ts) * (2 * DINNER) + d];
    }
    float sg = 1.f / (1.f + __expf(-acc));
    float r = acc * sg;
    xc[idx] = r;
    xc_h[idx] = (_Float16)r;
}

// ---------------------------------------------------------------
// Chunked selective scan (h_t = a_t h_{t-1} + b_t is associative).
// Pass 1: per (b,d,chunk) compute chunk summary (prod a = exp(A*sum dl), h_end).
// grid: 2 batches x 8 d-blocks x 16 chunks = 256 blocks, thread = channel.
__global__ __launch_bounds__(256)
void scan_pass1_kernel(const float* __restrict__ delta, const float* __restrict__ xc,
                       const float* __restrict__ proj, const float* __restrict__ A_log,
                       float* __restrict__ aprod, float* __restrict__ hend)
{
    __shared__ float Bs[CLEN * DSTATE];              // 16 KB
    const int tid   = threadIdx.x;
    const int gx    = blockIdx.x;
    const int bb    = gx >> 7;
    const int chunk = (gx >> 3) & 15;
    const int d     = ((gx & 7) << 8) + tid;
    const int tbase = chunk * CLEN;

    for (int i = tid; i < CLEN * DSTATE; i += 256) {
        int tt = i >> 4, s = i & 15;
        size_t tok = (size_t)bb * LSEQ + tbase + tt;
        Bs[i] = proj[tok * NPROJ + DTRANK + s];
    }
    __syncthreads();

    float Av[DSTATE], h[DSTATE];
#pragma unroll
    for (int s = 0; s < DSTATE; ++s) { Av[s] = -__expf(A_log[d * DSTATE + s]); h[s] = 0.f; }
    float dsum = 0.f;                                // sum of delta over chunk
    for (int tt = 0; tt < CLEN; ++tt) {
        size_t tok = (size_t)bb * LSEQ + tbase + tt;
        float dl = delta[tok * DINNER + d];
        float xv = xc[tok * DINNER + d];
        float dx = dl * xv;
        dsum += dl;
#pragma unroll
        for (int s = 0; s < DSTATE; ++s) {
            float dA = __expf(dl * Av[s]);
            h[s] = dA * h[s] + dx * Bs[tt * DSTATE + s];
        }
    }
    size_t base = ((size_t)bb * NCHUNK + chunk) * (DINNER * DSTATE) + (size_t)d * DSTATE;
#pragma unroll
    for (int s = 0; s < DSTATE; ++s) {
        aprod[base + s] = __expf(dsum * Av[s]);      // prod_t exp(dl_t*A_s) = exp(A_s * sum dl)
        hend[base + s]  = h[s];
    }
}

// Pass 2: tiny sequential prefix over the 16 chunk summaries, per (b,d,s).
__global__ __launch_bounds__(256)
void scan_prefix_kernel(const float* __restrict__ aprod, const float* __restrict__ hend,
                        float* __restrict__ hinit)
{
    int i = blockIdx.x * 256 + threadIdx.x;          // 2 * DINNER * DSTATE = 65536
    int bb = i >> 15;
    int ds = i & 32767;
    float h = 0.f;
    for (int c = 0; c < NCHUNK; ++c) {
        size_t idx = ((size_t)bb * NCHUNK + c) * (DINNER * DSTATE) + ds;
        hinit[idx] = h;
        h = aprod[idx] * h + hend[idx];
    }
}

// Pass 3: replay each chunk from corrected initial state; emit y with fused
// (+ xc*D) and (* silu(z)) epilogue, stored f16 for the out-GEMM.
__global__ __launch_bounds__(256)
void scan_pass3_kernel(const float* __restrict__ delta, const float* __restrict__ xc,
                       const float* __restrict__ xz, const float* __restrict__ proj,
                       const float* __restrict__ A_log, const float* __restrict__ Dp,
                       const float* __restrict__ hinit, _Float16* __restrict__ y_h)
{
    __shared__ float Bs[CLEN * DSTATE];
    __shared__ float Cs[CLEN * DSTATE];              // 32 KB total
    const int tid   = threadIdx.x;
    const int gx    = blockIdx.x;
    const int bb    = gx >> 7;
    const int chunk = (gx >> 3) & 15;
    const int d     = ((gx & 7) << 8) + tid;
    const int tbase = chunk * CLEN;

    for (int i = tid; i < CLEN * DSTATE; i += 256) {
        int tt = i >> 4, s = i & 15;
        size_t tok = (size_t)bb * LSEQ + tbase + tt;
        Bs[i] = proj[tok * NPROJ + DTRANK + s];
        Cs[i] = proj[tok * NPROJ + DTRANK + DSTATE + s];
    }
    __syncthreads();

    float Av[DSTATE], h[DSTATE];
    size_t hb = ((size_t)bb * NCHUNK + chunk) * (DINNER * DSTATE) + (size_t)d * DSTATE;
#pragma unroll
    for (int s = 0; s < DSTATE; ++s) {
        Av[s] = -__expf(A_log[d * DSTATE + s]);
        h[s]  = hinit[hb + s];
    }
    const float Dd = Dp[d];
    for (int tt = 0; tt < CLEN; ++tt) {
        size_t tok = (size_t)bb * LSEQ + tbase + tt;
        float dl = delta[tok * DINNER + d];
        float xv = xc[tok * DINNER + d];
        float zv = xz[tok * (2 * DINNER) + DINNER + d];
        float dx = dl * xv;
        float y = 0.f;
#pragma unroll
        for (int s = 0; s < DSTATE; ++s) {
            float dA = __expf(dl * Av[s]);
            h[s] = dA * h[s] + dx * Bs[tt * DSTATE + s];
            y += h[s] * Cs[tt * DSTATE + s];
        }
        float sg = 1.f / (1.f + __expf(-zv));
        y_h[tok * DINNER + d] = (_Float16)((y + xv * Dd) * (zv * sg));
    }
}

// ---------------------------------------------------------------
extern "C" void kernel_launch(void* const* d_in, const int* in_sizes, int n_in,
                              void* d_out, int out_size, void* d_ws, size_t ws_size,
                              hipStream_t stream)
{
    const float* x      = (const float*)d_in[0];
    const float* ln_w   = (const float*)d_in[1];
    const float* ln_b   = (const float*)d_in[2];
    const float* W_in   = (const float*)d_in[3];
    const float* b_in   = (const float*)d_in[4];
    const float* conv_w = (const float*)d_in[5];
    const float* conv_b = (const float*)d_in[6];
    const float* W_x    = (const float*)d_in[7];
    const float* W_dt   = (const float*)d_in[8];
    const float* b_dt   = (const float*)d_in[9];
    const float* A_log  = (const float*)d_in[10];
    const float* Dvec   = (const float*)d_in[11];
    const float* W_out  = (const float*)d_in[12];
    const float* b_out  = (const float*)d_in[13];
    float* out = (float*)d_out;

    // workspace carve-up (256B aligned slices)
    char* w = (char*)d_ws;
    size_t off = 0;
    auto take = [&](size_t bytes) { char* p = w + off; off += (bytes + 255) & ~(size_t)255; return p; };

    _Float16* xn_h    = (_Float16*)take((size_t)TOKENS * DMODEL * 2);
    _Float16* Win_h   = (_Float16*)take((size_t)(2 * DINNER) * DMODEL * 2);
    _Float16* Wx_h    = (_Float16*)take((size_t)NPROJ * DINNER * 2);
    _Float16* Wdt_h   = (_Float16*)take((size_t)DINNER * DTRANK * 2);
    _Float16* Wout_h  = (_Float16*)take((size_t)DMODEL * DINNER * 2);
    float*    xz      = (float*)take((size_t)TOKENS * 2 * DINNER * 4);
    float*    xc      = (float*)take((size_t)TOKENS * DINNER * 4);
    _Float16* xc_h    = (_Float16*)take((size_t)TOKENS * DINNER * 2);
    float*    proj    = (float*)take((size_t)TOKENS * NPROJ * 4);
    _Float16* dt_h    = (_Float16*)take((size_t)TOKENS * DTRANK * 2);
    float*    delta   = (float*)take((size_t)TOKENS * DINNER * 4);
    _Float16* y_h     = (_Float16*)take((size_t)TOKENS * DINNER * 2);
    const size_t SUMSZ = (size_t)2 * NCHUNK * DINNER * DSTATE * 4;   // 4 MB each
    float*    aprod   = (float*)take(SUMSZ);
    float*    hend    = (float*)take(SUMSZ);
    float*    hinit   = (float*)take(SUMSZ);
    (void)ws_size; (void)in_sizes; (void)n_in; (void)out_size;

    // 1) LayerNorm + cast
    ln_cast_kernel<<<TOKENS, 256, 0, stream>>>(x, ln_w, ln_b, xn_h);

    // 2) weight casts to f16
    auto cast = [&](const float* s, _Float16* dst, int n) {
        cast_h_kernel<<<(n + 255) / 256, 256, 0, stream>>>(s, dst, n);
    };
    cast(W_in,  Win_h,  2 * DINNER * DMODEL);
    cast(W_x,   Wx_h,   NPROJ * DINNER);
    cast(W_dt,  Wdt_h,  DINNER * DTRANK);
    cast(W_out, Wout_h, DMODEL * DINNER);

    // 3) xz = xn @ W_in^T + b_in   [8192 x 4096], K=1024
    {
        dim3 grid((2 * DINNER) / TILE_N, TOKENS / TILE_M);
        gemm_wmma_f16<0, false><<<grid, 256, 0, stream>>>(xn_h, Win_h, b_in, nullptr,
                                                          xz, TOKENS, 2 * DINNER, DMODEL);
    }

    // 4) causal depthwise conv + SiLU
    conv_silu_kernel<<<(TOKENS * DINNER) / 256, 256, 0, stream>>>(xz, conv_w, conv_b, xc, xc_h);

    // 5) proj = xc @ W_x^T   [8192 x 96], K=2048  (N not tile-aligned -> guarded)
    {
        dim3 grid((NPROJ + TILE_N - 1) / TILE_N, TOKENS / TILE_M);
        gemm_wmma_f16<0, true><<<grid, 256, 0, stream>>>(xc_h, Wx_h, nullptr, nullptr,
                                                         proj, TOKENS, NPROJ, DINNER);
    }

    // 6) dt (f16) from proj
    extract_dt_kernel<<<(TOKENS * DTRANK) / 256, 256, 0, stream>>>(proj, dt_h);

    // 7) delta = softplus(dt @ W_dt^T + b_dt)   [8192 x 2048], K=64
    {
        dim3 grid(DINNER / TILE_N, TOKENS / TILE_M);
        gemm_wmma_f16<1, false><<<grid, 256, 0, stream>>>(dt_h, Wdt_h, b_dt, nullptr,
                                                          delta, TOKENS, DINNER, DTRANK);
    }

    // 8) chunked selective scan: summaries -> prefix -> replay (+ gate epilogue)
    scan_pass1_kernel<<<256, 256, 0, stream>>>(delta, xc, proj, A_log, aprod, hend);
    scan_prefix_kernel<<<256, 256, 0, stream>>>(aprod, hend, hinit);
    scan_pass3_kernel<<<256, 256, 0, stream>>>(delta, xc, xz, proj, A_log, Dvec,
                                               hinit, y_h);

    // 9) out = x + y @ W_out^T + b_out   [8192 x 1024], K=2048
    {
        dim3 grid(DMODEL / TILE_N, TOKENS / TILE_M);
        gemm_wmma_f16<2, false><<<grid, 256, 0, stream>>>(y_h, Wout_h, b_out, x,
                                                          out, TOKENS, DMODEL, DINNER);
    }
}
